// FisheyeConv_46849503265181
// MI455X (gfx1250) — matrix operand order repaired and checked by reading, not compiled, not run on hardware
//
#include <hip/hip_runtime.h>
#include <hip/hip_bf16.h>

// ---------------------------------------------------------------------------
// FisheyeConv for MI455X (gfx1250, wave32, WMMA).
//
// Algebraic collapse: nb_vals is independent of both group indices (l, l'),
// and pool/linear are linear ops, so the per-(p, kL) MLP weights reduce to
//   s_bar[i][ki][m] = (1/L) * sum_{l,l'} pw[(i,l)][(ki,l')][m] * valid(i,ki)
// giving:
//   pooled[i][o] = (V_i^T(64x29) @ s_bar_i(29x16)).flat(1024) @ Wlin + blin
// followed by BN + SiLU.  Contractions run on v_wmma_f32_16x16x32_bf16.
//
// All intermediates are stored directly in WMMA fragment order so operand
// construction is pure b128 loads (the GEMM-K permutation is applied
// identically to P and to the pre-converted Wlin, so results are unchanged).
//
// Workspace layout (requires ws_size >= 10 MB):
//   [0    , 288KB) : xb  bf16[2304][64]    pixel-major activations
//   [1MB  , 3.25MB): S   bf16[2304][512]   reduced MLP weights, B-frag order
//   [4MB  , 8.5MB) : P   bf16[2304][1024]  per-pixel partials, C-frag order
//   [9MB  , 9.13MB): Wt  bf16[64][1024]    Wlin, transposed + K-permuted
// ---------------------------------------------------------------------------

typedef __attribute__((ext_vector_type(16))) __bf16 v16bf;
typedef __attribute__((ext_vector_type(8)))  float  v8f;
typedef int v4i_gcc __attribute__((vector_size(16)));   // matches builtin param

#define NPIX   2304
#define HW     48
#define NOFF   29
#define CCH    64
#define CM     16
#define KPAD   32

#if defined(__AMDGCN__) && __has_builtin(__builtin_amdgcn_global_load_async_to_lds_b128) && __has_builtin(__builtin_amdgcn_s_wait_asynccnt)
#define USE_ASYNC 1
#else
#define USE_ASYNC 0
#endif

__constant__ int OFF_DX[NOFF] = { 0,
                                  -2,-1,0,1,2,
                                  -2,-1,0,1,2,
                                  -3,-2,-1,0,1,2,3,
                                  -2,-1,0,1,2,
                                  -2,-1,0,1,2,
                                  0 };
__constant__ int OFF_DY[NOFF] = { -3,
                                  -2,-2,-2,-2,-2,
                                  -1,-1,-1,-1,-1,
                                   0, 0, 0,0,0,0,0,
                                   1, 1, 1,1,1,
                                   2, 2, 2,2,2,
                                   3 };

__device__ __forceinline__ unsigned short f32_to_bf16(float f) {
    union { float f; unsigned u; } v; v.f = f;
    unsigned u = v.u;
    return (unsigned short)((u + 0x7FFFu + ((u >> 16) & 1u)) >> 16);  // RNE
}
__device__ __forceinline__ __bf16 us_to_bf(unsigned short s) {
    union { unsigned short u; __bf16 b; } v; v.u = s; return v.b;
}
__device__ __forceinline__ float silu(float x) {
    return x / (1.0f + __expf(-x));
}

// ----- SO(3) helpers (closed-form, fp32) -----------------------------------
__device__ __forceinline__ void so3_exp3(float x, float y, float z, float R[9]) {
    float t2 = x * x + y * y + z * z;
    float th = sqrtf(t2);
    float A, B;
    if (th < 1e-6f) { A = 1.0f - t2 * (1.0f / 6.0f); B = 0.5f - t2 * (1.0f / 24.0f); }
    else            { A = sinf(th) / th;             B = (1.0f - cosf(th)) / t2;      }
    R[0] = 1.0f - B * (y * y + z * z); R[1] = -A * z + B * x * y;          R[2] =  A * y + B * x * z;
    R[3] =  A * z + B * x * y;         R[4] = 1.0f - B * (x * x + z * z);  R[5] = -A * x + B * y * z;
    R[6] = -A * y + B * x * z;         R[7] =  A * x + B * y * z;          R[8] = 1.0f - B * (x * x + y * y);
}

__device__ __forceinline__ void so3_log3(const float R[9], float v[3]) {
    float tr  = R[0] + R[4] + R[8];
    float c   = fminf(fmaxf((tr - 1.0f) * 0.5f, -1.0f), 1.0f);
    float th  = acosf(c);
    float s   = sinf(th);
    float fac = (s < 1e-5f) ? 0.5f : th / (2.0f * s);
    v[0] = fac * (R[7] - R[5]);
    v[1] = fac * (R[2] - R[6]);
    v[2] = fac * (R[3] - R[1]);
}

// base fisheye-lift rotation R0(i) for pixel i
__device__ __forceinline__ void base_rot(int i, float R[9]) {
    float u  = (float)(i % HW), v = (float)(i / HW);
    float du = u - 23.5f, dv = v - 23.5f;
    float r  = sqrtf(du * du + dv * dv);
    float th = r * (1.0f / 24.0f);
    bool  onax = r < 1e-8f;
    float sr = onax ? 1.0f : r;
    float sth = sinf(th);
    float rx = onax ? 0.0f : sth * du / sr;
    float ry = onax ? 0.0f : sth * dv / sr;
    float rz = cosf(th);
    float sinn = sqrtf(rx * rx + ry * ry);
    float ang  = acosf(fminf(fmaxf(rz, -1.0f), 1.0f));
    float fac  = (sinn < 1e-8f) ? 0.0f : ang / sinn;
    so3_exp3(-fac * ry, fac * rx, 0.0f, R);
}

// ----- K0: x (64,2304) f32 -> xb (2304,64) bf16 -----------------------------
__global__ void k0_convert(const float* __restrict__ x, unsigned short* __restrict__ xb) {
    int t   = blockIdx.x * 256 + threadIdx.x;   // 147456 threads exactly
    int pix = t >> 6, c = t & 63;
    xb[pix * CCH + c] = f32_to_bf16(x[c * NPIX + pix]);
}

// ----- K0b: Wlin (1024,64) f32 -> Wt (64,1024) bf16, K permuted to match P --
// store index f = t4*256 + lane*8 + r  <->  logical feature (crow*16 + m)
// with crow = t4*16 + r + ((lane&16)?8:0), m = lane&15   (C-fragment order)
__global__ void k0b_wlin(const float* __restrict__ Wlin, unsigned short* __restrict__ Wt) {
    int t = blockIdx.x * 256 + threadIdx.x;     // 65536 threads exactly
    int o = t >> 10, f = t & 1023;
    int t4 = f >> 8, lane = (f >> 3) & 31, r = f & 7;
    int crow = t4 * 16 + r + ((lane & 16) ? 8 : 0);
    int m    = lane & 15;
    Wt[o * 1024 + f] = f32_to_bf16(Wlin[(crow * CM + m) * 64 + o]);
}

// ----- K1: geometry MLP, reduced over (l,l'); S written in B-frag order -----
__global__ void k1_geom_mlp(const float* __restrict__ W1, const float* __restrict__ b1,
                            const float* __restrict__ W2, const float* __restrict__ b2,
                            const float* __restrict__ W3, const float* __restrict__ b3,
                            unsigned short* __restrict__ S) {
    __shared__ float sW1[96], sB1[32], sW2[1024], sB2[32], sW3[512], sB3[16];
    for (int t = threadIdx.x; t < 96;   t += 256) sW1[t] = W1[t];
    for (int t = threadIdx.x; t < 32;   t += 256) { sB1[t] = b1[t]; sB2[t] = b2[t]; }
    for (int t = threadIdx.x; t < 1024; t += 256) sW2[t] = W2[t];
    for (int t = threadIdx.x; t < 512;  t += 256) sW3[t] = W3[t];
    for (int t = threadIdx.x; t < 16;   t += 256) sB3[t] = b3[t];
    __syncthreads();

    int t  = blockIdx.x * 256 + threadIdx.x;    // 2304*32 threads exactly
    int i  = t >> 5, ki = t & 31;
    int y  = i / HW, x = i % HW;

    float acc[CM];
    #pragma unroll
    for (int m = 0; m < CM; ++m) acc[m] = 0.0f;

    bool valid = false;
    int  i2    = i;
    if (ki < NOFF) {
        int nx = x + OFF_DX[ki], ny = y + OFF_DY[ki];
        valid = (nx >= 0) & (nx < HW) & (ny >= 0) & (ny < HW);
        if (valid) i2 = ny * HW + nx;
    }

    if (valid) {
        float R0i[9], R0n[9], M[9];
        base_rot(i,  R0i);
        base_rot(i2, R0n);
        // M = R0n^T * R0i
        #pragma unroll
        for (int r = 0; r < 3; ++r)
            #pragma unroll
            for (int c = 0; c < 3; ++c)
                M[r * 3 + c] = R0n[0 * 3 + r] * R0i[0 * 3 + c]
                             + R0n[1 * 3 + r] * R0i[1 * 3 + c]
                             + R0n[2 * 3 + r] * R0i[2 * 3 + c];

        const float CA[4] = { 1.0f, 0.0f, -1.0f,  0.0f };
        const float SA[4] = { 0.0f, 1.0f,  0.0f, -1.0f };

        for (int l = 0; l < 4; ++l) {
            float cl = CA[l], sl = SA[l];
            float T[9];                                   // T = M * Rz(l)
            #pragma unroll
            for (int r = 0; r < 3; ++r) {
                T[r * 3 + 0] =  M[r * 3 + 0] * cl + M[r * 3 + 1] * sl;
                T[r * 3 + 1] = -M[r * 3 + 0] * sl + M[r * 3 + 1] * cl;
                T[r * 3 + 2] =  M[r * 3 + 2];
            }
            for (int l2 = 0; l2 < 4; ++l2) {
                float c2 = CA[l2], s2 = SA[l2];
                float Rr[9];                              // Rr = Rz(l2)^T * T
                #pragma unroll
                for (int c = 0; c < 3; ++c) {
                    Rr[0 * 3 + c] =  c2 * T[0 * 3 + c] + s2 * T[1 * 3 + c];
                    Rr[1 * 3 + c] = -s2 * T[0 * 3 + c] + c2 * T[1 * 3 + c];
                    Rr[2 * 3 + c] =  T[2 * 3 + c];
                }
                float ab[3];
                so3_log3(Rr, ab);
                // MLP 3 -> 32 -> 32 -> 16 (fp32)
                float h1[32];
                #pragma unroll 4
                for (int j = 0; j < 32; ++j)
                    h1[j] = silu(ab[0] * sW1[j] + ab[1] * sW1[32 + j]
                               + ab[2] * sW1[64 + j] + sB1[j]);
                float h2[32];
                for (int j = 0; j < 32; ++j) {
                    float s = sB2[j];
                    #pragma unroll 8
                    for (int q = 0; q < 32; ++q) s += h1[q] * sW2[q * 32 + j];
                    h2[j] = silu(s);
                }
                #pragma unroll
                for (int m = 0; m < CM; ++m) {
                    float s = sB3[m];
                    #pragma unroll 8
                    for (int q = 0; q < 32; ++q) s += h2[q] * sW3[q * CM + m];
                    acc[m] += s;
                }
            }
        }
    }
    // B-fragment order: slot(lane = m + (ki&16), e = ki&15) holds s_bar[ki][m]
    int base = i * (KPAD * CM) + (ki & 16) * 16 + (ki & 15);
    #pragma unroll
    for (int m = 0; m < CM; ++m)
        S[base + m * 16] = f32_to_bf16(acc[m] * 0.25f);
}

// ----- K2: per-pixel gather GEMM  P_i(64x16) = V_i^T(64x32) @ S_i(32x16) ----
__global__ void k2_gather_gemm(const unsigned short* __restrict__ xb,
                               const unsigned short* __restrict__ S,
                               unsigned short* __restrict__ P) {
    __shared__ unsigned short Vlds[8][KPAD * CCH];   // 32 rows x 64 ch, per wave

    int wv   = threadIdx.x >> 5;
    int lane = threadIdx.x & 31;
    int i    = blockIdx.x * 8 + wv;                  // 288 blocks * 8 waves = 2304
    int y    = i / HW, x = i % HW;

    // stage V: lane owns row ki = lane (invalid / pad rows get zero weight in S)
    int i2 = i;
    if (lane < NOFF) {
        int nx = x + OFF_DX[lane], ny = y + OFF_DY[lane];
        if (nx >= 0 && nx < HW && ny >= 0 && ny < HW) i2 = ny * HW + nx;
    }
    {
        const uint4* src = (const uint4*)(xb + i2 * CCH);
        uint4*       dst = (uint4*)(&Vlds[wv][lane * CCH]);
#if USE_ASYNC
        #pragma unroll
        for (int q = 0; q < 8; ++q)
            __builtin_amdgcn_global_load_async_to_lds_b128(
                (v4i_gcc*)(src + q), (v4i_gcc*)(dst + q), 0, 0);
        __builtin_amdgcn_s_wait_asynccnt(0);
#else
        #pragma unroll
        for (int q = 0; q < 8; ++q) dst[q] = src[q];
#endif
    }
    __syncthreads();

    // B fragment: stored in fragment order -> two contiguous b128 loads
    union { v16bf v; unsigned u[8]; } Bu;
    {
        const uint4* sp = (const uint4*)(S + (size_t)i * (KPAD * CM) + lane * 16);
        *(uint4*)&Bu.u[0] = sp[0];
        *(uint4*)&Bu.u[4] = sp[1];
    }

    int mlo  = lane & 15;
    int krow = (lane & 16) ? 8 : 0;
    #pragma unroll
    for (int t4 = 0; t4 < 4; ++t4) {                 // 4 M-tiles over 64 channels
        v16bf afr;
        #pragma unroll
        for (int e = 0; e < 16; ++e) {
            int K = krow * 2 + e;
            afr[e] = us_to_bf(Vlds[wv][K * CCH + t4 * 16 + mlo]);
        }
        v8f accv = {};
        accv = __builtin_amdgcn_wmma_f32_16x16x32_bf16(
                   false, afr, false, Bu.v, (short)0, accv, false, false);
        // store raw C fragment: 8 bf16 per lane, one b128 store
        uint4 st;
        st.x = (unsigned)f32_to_bf16(accv[0]) | ((unsigned)f32_to_bf16(accv[1]) << 16);
        st.y = (unsigned)f32_to_bf16(accv[2]) | ((unsigned)f32_to_bf16(accv[3]) << 16);
        st.z = (unsigned)f32_to_bf16(accv[4]) | ((unsigned)f32_to_bf16(accv[5]) << 16);
        st.w = (unsigned)f32_to_bf16(accv[6]) | ((unsigned)f32_to_bf16(accv[7]) << 16);
        *(uint4*)(P + (size_t)i * 1024 + t4 * 256 + lane * 8) = st;
    }
}

// ----- K3: (2304 x 1024) @ (1024 x 64) + blin, BN, SiLU, transpose-out ------
__global__ void k3_lin_bn(const unsigned short* __restrict__ P,
                          const unsigned short* __restrict__ Wt,
                          const float* __restrict__ blin,
                          const float* __restrict__ gamma, const float* __restrict__ beta,
                          const float* __restrict__ rmean, const float* __restrict__ rvar,
                          float* __restrict__ out) {
    int wv   = threadIdx.x >> 5;
    int lane = threadIdx.x & 31;
    int wid  = blockIdx.x * 8 + wv;                  // 72 blocks * 8 = 576 tiles
    int mt   = wid >> 2;                             // 144 row tiles
    int nt   = wid & 3;                              // 4 col tiles

    int irow = mt * 16 + (lane & 15);                // A row (pixel)
    int o    = nt * 16 + (lane & 15);                // B col (output channel)
    int krow = (lane & 16) ? 8 : 0;
    int kB   = (lane & 16) ? 16 : 0;

    const unsigned* arow = (const unsigned*)(P  + (size_t)irow * 1024);
    const unsigned* brow = (const unsigned*)(Wt + (size_t)o    * 1024);

    v8f accv = {};
    for (int kk = 0; kk < 1024; kk += 32) {
        __builtin_prefetch(P + (size_t)irow * 1024 + kk + 256, 0, 1);
        union { v16bf v; unsigned u[8]; } Au, Bw;
        int ad = (kk + krow) >> 1;                   // dword base for A
        #pragma unroll
        for (int vq = 0; vq < 4; ++vq) {
            Au.u[vq]     = arow[ad + vq];            // K = krow + {0..7}
            Au.u[vq + 4] = arow[ad + 8 + vq];        // K = 16 + krow + {0..7}
        }
        int bd = (kk + krow) >> 1;                   // dword base for B
        #pragma unroll
        for (int vq = 0; vq < 8; ++vq)
            Bw.u[vq] = brow[bd + vq + (vq & 4)];     // K = krow + {0..7}, 16 + krow + {0..7}
        accv = __builtin_amdgcn_wmma_f32_16x16x32_bf16(
                   false, Au.v, false, Bw.v, (short)0, accv, false, false);
    }

    float bl  = blin[o];
    float inv = rsqrtf(rvar[o] + 1e-5f);
    float g   = gamma[o], bb = beta[o], rm = rmean[o];
    #pragma unroll
    for (int r = 0; r < 8; ++r) {
        int   i  = mt * 16 + r + ((lane & 16) ? 8 : 0);
        float v  = accv[r] + bl;
        float yv = (v - rm) * inv * g + bb;
        out[o * NPIX + i] = silu(yv);
    }
}

// ---------------------------------------------------------------------------
extern "C" void kernel_launch(void* const* d_in, const int* in_sizes, int n_in,
                              void* d_out, int out_size, void* d_ws, size_t ws_size,
                              hipStream_t stream) {
    (void)in_sizes; (void)n_in; (void)out_size; (void)ws_size;
    const float* x     = (const float*)d_in[0];
    const float* W1    = (const float*)d_in[1];
    const float* b1    = (const float*)d_in[2];
    const float* W2    = (const float*)d_in[3];
    const float* b2    = (const float*)d_in[4];
    const float* W3    = (const float*)d_in[5];
    const float* b3    = (const float*)d_in[6];
    const float* Wlin  = (const float*)d_in[7];
    const float* blin  = (const float*)d_in[8];
    const float* gamma = (const float*)d_in[9];
    const float* beta  = (const float*)d_in[10];
    const float* rmean = (const float*)d_in[11];
    const float* rvar  = (const float*)d_in[12];

    char* ws = (char*)d_ws;
    unsigned short* xb = (unsigned short*)(ws);                    // 288 KB
    unsigned short* S  = (unsigned short*)(ws + (1u << 20));       // 2.25 MB
    unsigned short* P  = (unsigned short*)(ws + (4u << 20));       // 4.5 MB
    unsigned short* Wt = (unsigned short*)(ws + (9u << 20));       // 128 KB

    k0_convert    <<<576, 256, 0, stream>>>(x, xb);
    k0b_wlin      <<<256, 256, 0, stream>>>(Wlin, Wt);
    k1_geom_mlp   <<<288, 256, 0, stream>>>(W1, b1, W2, b2, W3, b3, S);
    k2_gather_gemm<<<288, 256, 0, stream>>>(xb, S, P);
    k3_lin_bn     <<<72, 256, 0, stream>>>(P, Wt, blin, gamma, beta, rmean, rvar,
                                           (float*)d_out);
}
